// MSGClassification_55972013801667
// MI455X (gfx1250) — compile-verified, matrix operand order checked by reading, not run on hardware
//
#include <hip/hip_runtime.h>
#include <hip/hip_bf16.h>

typedef __bf16 bf16_t;
typedef __attribute__((ext_vector_type(16))) __bf16 v16bf;
typedef __attribute__((ext_vector_type(8)))  float  v8f;

// ---------- bf16 helpers: native conversions (clang __bf16 arithmetic type) ----------
__device__ __forceinline__ bf16_t f2bf(float f) { return static_cast<bf16_t>(f); }
__device__ __forceinline__ float  bf2f(bf16_t b) { return static_cast<float>(b); }

// K index map inside a 16-bit A/B fragment (CDNA5 ISA 7.12.2):
// VGPR r (pair j): r<4 -> k=2r+j ; r>=4 -> k=16+2(r-4)+j ; lanes 16..31 add +8.
__device__ __forceinline__ int kmap(int i) {
    int r = i >> 1, j = i & 1;
    return (r < 4) ? (r * 2 + j) : (16 + (r - 4) * 2 + j);
}

__device__ __forceinline__ v8f wmma_bf16(v16bf a, v16bf b, v8f c) {
    return __builtin_amdgcn_wmma_f32_16x16x32_bf16(false, a, false, b, (short)0, c, false, false);
}

// A fragment from row-major X (ld elements per row), rows = lane&15, k window 32 wide.
// Per-lane data is two contiguous 8-element runs -> lowers to 2x b128 loads.
__device__ __forceinline__ v16bf load_a(const bf16_t* X, int ld, int lane) {
    int m = lane & 15, half = lane >> 4;
    const bf16_t* row = X + (size_t)m * ld + half * 8;
    v16bf a;
#pragma unroll
    for (int i = 0; i < 16; ++i) a[i] = row[kmap(i)];
    return a;
}
// B fragment from K-major W (ldn elements per k-row): element (k, n0 + (lane&15)).
// Each of the 16 loads is a coalesced 2B/lane access across lanes.
__device__ __forceinline__ v16bf load_b(const bf16_t* Wp, int ldn, int lane) {
    int half = lane >> 4;
    const bf16_t* cp = Wp + (lane & 15) + (size_t)(half * 8) * ldn;
    v16bf b;
#pragma unroll
    for (int i = 0; i < 16; ++i) b[i] = cp[(size_t)kmap(i) * ldn];
    return b;
}

// ---------- small utility kernels ----------
__global__ void transpose_x_kernel(const float* __restrict__ x, float* __restrict__ xyz) {
    int e = blockIdx.x * 256 + threadIdx.x;            // B*N
    if (e >= 16 * 4096) return;
    int b = e / 4096, p = e % 4096;
#pragma unroll
    for (int d = 0; d < 3; ++d)
        xyz[(size_t)e * 3 + d] = x[((size_t)b * 3 + d) * 4096 + p];
}

__global__ __launch_bounds__(256) void fps_kernel(const float* __restrict__ pts, int n, int npoint,
                                                  int* __restrict__ out) {
    __shared__ float dist[4096];
    __shared__ float sv[256];
    __shared__ int   si[256];
    __shared__ int   sFar;
    int b = blockIdx.x, tid = threadIdx.x;
    const float* P = pts + (size_t)b * n * 3;
    for (int i = tid; i < n; i += 256) dist[i] = 1e10f;
    if (tid == 0) sFar = 0;
    __syncthreads();
    for (int it = 0; it < npoint; ++it) {
        int far = sFar;
        if (tid == 0) out[b * npoint + it] = far;
        float cx = P[far * 3], cy = P[far * 3 + 1], cz = P[far * 3 + 2];
        float bv = -1.0f; int bi = 0;
        for (int i = tid; i < n; i += 256) {
            float dx = P[i * 3] - cx, dy = P[i * 3 + 1] - cy, dz = P[i * 3 + 2] - cz;
            float d = dx * dx + dy * dy + dz * dz;
            float dd = fminf(dist[i], d);
            dist[i] = dd;
            if (dd > bv) { bv = dd; bi = i; }
        }
        sv[tid] = bv; si[tid] = bi;
        __syncthreads();
        for (int o = 128; o; o >>= 1) {
            if (tid < o) {
                if (sv[tid + o] > sv[tid] ||
                    (sv[tid + o] == sv[tid] && si[tid + o] < si[tid])) {
                    sv[tid] = sv[tid + o]; si[tid] = si[tid + o];
                }
            }
            __syncthreads();
        }
        if (tid == 0) sFar = si[0];
        __syncthreads();
    }
}

__global__ void gather3_kernel(const float* __restrict__ src, int Ns,
                               const int* __restrict__ idx, int P, float* __restrict__ dst) {
    int e = blockIdx.x * 256 + threadIdx.x;            // B*P
    if (e >= 16 * P) return;
    int b = e / P;
    int j = idx[e];
#pragma unroll
    for (int d = 0; d < 3; ++d)
        dst[(size_t)e * 3 + d] = src[((size_t)b * Ns + j) * 3 + d];
}

// one wave per centroid: first-50-in-ball (by index), pad to 64 with first hit
__global__ __launch_bounds__(128) void ballq_kernel(const float* __restrict__ xyz,
                                                    const float* __restrict__ nxyz,
                                                    int Npts, int P, float r2,
                                                    int* __restrict__ idxg) {
    int wid = blockIdx.x * (blockDim.x >> 5) + (threadIdx.x >> 5);
    int lane = threadIdx.x & 31;
    if (wid >= 16 * P) return;
    int b = wid / P;
    const float* c = nxyz + (size_t)wid * 3;
    float cx = c[0], cy = c[1], cz = c[2];
    int* out = idxg + (size_t)wid * 64;
    const float* pb = xyz + (size_t)b * Npts * 3;
    int cnt = 0; int firstj = 0x7fffffff;
    for (int base = 0; base < Npts && cnt < 50; base += 32) {
        int j = base + lane;
        bool inr = false;
        if (j < Npts) {
            float dx = pb[j * 3] - cx, dy = pb[j * 3 + 1] - cy, dz = pb[j * 3 + 2] - cz;
            inr = (dx * dx + dy * dy + dz * dz) <= r2;
        }
        unsigned long long m = __ballot(inr);
        int pos = (int)__popcll(m & ((1ull << lane) - 1ull));
        if (inr) {
            int slot = cnt + pos;
            if (slot < 50) { out[slot] = j; if (slot == 0) firstj = j; }
        }
        cnt += (int)__popcll(m);
    }
    for (int o = 16; o; o >>= 1) {
        int t = __shfl_down(firstj, o, 32);
        firstj = (t < firstj) ? t : firstj;
    }
    firstj = __shfl(firstj, 0, 32);
    if (firstj == 0x7fffffff) firstj = 0;
    int c0 = cnt < 50 ? cnt : 50;
    for (int t = c0 + lane; t < 64; t += 32) out[t] = firstj;
}

// Weight prep: fold BN scale into W, produce K-major bf16 [KP][NP], bias float[NP]
__global__ void prep_w_kernel(const float* __restrict__ W, const float* __restrict__ bvec,
                              const float* __restrict__ g, const float* __restrict__ bt,
                              int cin, int cout, int KP, int NP, float s,
                              bf16_t* __restrict__ wo, float* __restrict__ bo) {
    int e = blockIdx.x * 256 + threadIdx.x;
    if (e < KP * NP) {
        int k = e / NP, n = e % NP;
        float v = 0.f;
        if (k < cin && n < cout) {
            float sc = g ? g[n] * s : 1.f;
            v = W[(size_t)n * cin + k] * sc;
        }
        wo[e] = f2bf(v);
    }
    if (e < NP) {
        int n = e;
        float bb = 0.f;
        if (n < cout) {
            float sc = g ? g[n] * s : 1.f;
            bb = (bvec ? bvec[n] * sc : 0.f) + (bt ? bt[n] : 0.f);
        }
        bo[n] = bb;
    }
}

// ---------- fused grouped MLP layers (64 rows resident in LDS) ----------
template <int K, int COUT>
__device__ __forceinline__ void layer_lds(const bf16_t* X, const bf16_t* W, const float* bias,
                                          bf16_t* Y, int wave, int lane) {
    constexpr int NT = COUT / 16;
    for (int t = wave; t < 4 * NT; t += 4) {
        int mt = t / NT, nt = t % NT;
        v8f acc = {};
#pragma unroll
        for (int k = 0; k < K; k += 32) {
            v16bf a = load_a(X + (size_t)mt * 16 * K + k, K, lane);
            v16bf b = load_b(W + (size_t)k * COUT + nt * 16, COUT, lane);
            acc = wmma_bf16(a, b, acc);
        }
        int n = nt * 16 + (lane & 15), half = lane >> 4;
        float bs = bias[n];
#pragma unroll
        for (int r = 0; r < 8; ++r) {
            int m = mt * 16 + r + 8 * half;
            float v = fmaxf(acc[r] + bs, 0.f);
            Y[(size_t)m * COUT + n] = f2bf(v);
        }
    }
}

// Layer 1 for SA2: gather grouped features (f1 + relative xyz) on the fly
template <int K, int COUT>
__device__ __forceinline__ void layer_gather(const int* sIdx, const float* sRel,
                                             const float* feat, int FC, size_t browoff,
                                             const bf16_t* W, const float* bias,
                                             bf16_t* Y, int wave, int lane) {
    constexpr int NT = COUT / 16;
    for (int t = wave; t < 4 * NT; t += 4) {
        int mt = t / NT, nt = t % NT;
        int s = mt * 16 + (lane & 15);
        int j = sIdx[s];
        const float* frow = feat + (browoff + (size_t)j) * FC;
        int half = lane >> 4;
        v8f acc = {};
#pragma unroll
        for (int k0 = 0; k0 < K; k0 += 32) {
            v16bf a;
#pragma unroll
            for (int i = 0; i < 16; ++i) {
                int k = k0 + half * 8 + kmap(i);
                float v;
                if (k < FC)            v = frow[k];
                else if (k < FC + 3)   v = sRel[s * 3 + (k - FC)];
                else                   v = 0.f;
                a[i] = f2bf(v);
            }
            v16bf b = load_b(W + (size_t)k0 * COUT + nt * 16, COUT, lane);
            acc = wmma_bf16(a, b, acc);
        }
        int n = nt * 16 + (lane & 15);
        float bs = bias[n];
#pragma unroll
        for (int r = 0; r < 8; ++r) {
            int m = mt * 16 + r + 8 * half;
            float v = fmaxf(acc[r] + bs, 0.f);
            Y[(size_t)m * COUT + n] = f2bf(v);
        }
    }
}

template <int K, int COUT>
__device__ __forceinline__ void layer_pool(const bf16_t* X, const bf16_t* W, const float* bias,
                                           float* pool, int wave, int lane) {
    constexpr int NT = COUT / 16;
    for (int t = wave; t < 4 * NT; t += 4) {
        int mt = t / NT, nt = t % NT;
        v8f acc = {};
#pragma unroll
        for (int k = 0; k < K; k += 32) {
            v16bf a = load_a(X + (size_t)mt * 16 * K + k, K, lane);
            v16bf b = load_b(W + (size_t)k * COUT + nt * 16, COUT, lane);
            acc = wmma_bf16(a, b, acc);
        }
        int n = nt * 16 + (lane & 15);
        float bs = bias[n];
        float mx = 0.f;
#pragma unroll
        for (int r = 0; r < 8; ++r) mx = fmaxf(mx, acc[r] + bs);
        mx = fmaxf(mx, 0.f);
        atomicMax((int*)&pool[n], __float_as_int(mx));   // relu => nonneg, int-max == float-max
    }
}

// MODE 0: SA1 (input = relative xyz only); MODE 1: SA2 (feat + relative xyz)
template <int CINP, int C1, int C2, int C3, int MODE>
__global__ __launch_bounds__(128) void fused_branch_kernel(
    const float* __restrict__ xyz, const float* __restrict__ nxyz,
    const int* __restrict__ idxg, const float* __restrict__ feat, int FC,
    int Npts, int P,
    const bf16_t* W1, const float* B1,
    const bf16_t* W2, const float* B2,
    const bf16_t* W3, const float* B3,
    float* __restrict__ outF, int FOUT, int coff) {
    __shared__ int    sIdx[64];
    __shared__ float  sRel[64 * 3];
    __shared__ bf16_t sIn[(MODE == 0) ? (64 * CINP) : 16];
    __shared__ bf16_t sH1[64 * C1];
    __shared__ bf16_t sH2[64 * C2];
    __shared__ float  sPool[C3];

    int bp = blockIdx.x;
    int b = bp / P;
    int tid = threadIdx.x, lane = tid & 31, wave = tid >> 5;
    const float* cc = nxyz + (size_t)bp * 3;
    float cx = cc[0], cy = cc[1], cz = cc[2];
    const int* ig = idxg + (size_t)bp * 64;
    if (tid < 64) sIdx[tid] = ig[tid];
    __syncthreads();
    if (tid < 64) {
        int j = sIdx[tid];
        const float* q = xyz + ((size_t)b * Npts + j) * 3;
        sRel[tid * 3 + 0] = q[0] - cx;
        sRel[tid * 3 + 1] = q[1] - cy;
        sRel[tid * 3 + 2] = q[2] - cz;
    }
    for (int c = tid; c < C3; c += 128) sPool[c] = 0.f;
    if constexpr (MODE == 0) {
        __syncthreads();
        for (int e = tid; e < 64 * CINP; e += 128) {
            int s = e / CINP, c = e % CINP;
            float v = (c < 3) ? sRel[s * 3 + c] : 0.f;
            sIn[e] = f2bf(v);
        }
    }
    __syncthreads();
    if constexpr (MODE == 0) {
        layer_lds<CINP, C1>(sIn, W1, B1, sH1, wave, lane);
    } else {
        layer_gather<CINP, C1>(sIdx, sRel, feat, FC, (size_t)b * Npts, W1, B1, sH1, wave, lane);
    }
    __syncthreads();
    layer_lds<C1, C2>(sH1, W2, B2, sH2, wave, lane);
    __syncthreads();
    layer_pool<C2, C3>(sH2, W3, B3, sPool, wave, lane);
    __syncthreads();
    float* of = outF + (size_t)bp * FOUT + coff;
    for (int c = tid; c < C3; c += 128) of[c] = sPool[c];
}

// ---------- generic WMMA GEMM (SA3 + FC): Y = relu?(X @ W + bias) ----------
// NT = compile-time number of 16-col tiles in this block's 64-col strip so the
// accumulator array is statically indexed (no v_movrel around the WMMAs).
template <int NT>
__global__ __launch_bounds__(128) void gemm_wmma_kernel(
    const bf16_t* __restrict__ X, int M, int KP,
    const bf16_t* __restrict__ W, int NP, const float* __restrict__ bias,
    bf16_t* __restrict__ Ybf, float* __restrict__ Yf, int relu) {
    int lane = threadIdx.x & 31, wave = threadIdx.x >> 5;
    int m0 = blockIdx.x * 64 + wave * 16;
    int n0 = blockIdx.y * 64;
    int half = lane >> 4;
    int mrow = m0 + (lane & 15);
    bool mv = mrow < M;
    const bf16_t* xr = X + (size_t)mrow * KP;
    v8f acc[NT];
#pragma unroll
    for (int nt = 0; nt < NT; ++nt) acc[nt] = (v8f){};
    for (int k = 0; k < KP; k += 32) {
        if (k + 32 < KP) __builtin_prefetch(W + (size_t)(k + 32) * NP + n0, 0, 0);
        v16bf a;
        if (mv) {
            const bf16_t* rp = xr + k + half * 8;
#pragma unroll
            for (int i = 0; i < 16; ++i) a[i] = rp[kmap(i)];
        } else {
            bf16_t z = f2bf(0.f);
#pragma unroll
            for (int i = 0; i < 16; ++i) a[i] = z;
        }
#pragma unroll
        for (int nt = 0; nt < NT; ++nt) {
            v16bf b = load_b(W + (size_t)k * NP + n0 + nt * 16, NP, lane);
            acc[nt] = wmma_bf16(a, b, acc[nt]);
        }
    }
#pragma unroll
    for (int nt = 0; nt < NT; ++nt) {
        int n = n0 + nt * 16 + (lane & 15);
        float bs = bias[n];
#pragma unroll
        for (int r = 0; r < 8; ++r) {
            int m = m0 + r + 8 * half;
            if (m >= M) continue;
            float v = acc[nt][r] + bs;
            if (relu) v = fmaxf(v, 0.f);
            if (Yf) Yf[(size_t)m * NP + n] = v;
            else    Ybf[(size_t)m * NP + n] = f2bf(v);
        }
    }
}

// SA3 input: [xyz2 | f2] padded 643 -> 672, bf16, rows = B*128
__global__ void build_g3_kernel(const float* __restrict__ xyz2, const float* __restrict__ f2,
                                bf16_t* __restrict__ g3) {
    int e = blockIdx.x * 256 + threadIdx.x;
    if (e >= 2048 * 672) return;
    int r = e / 672, c = e % 672;
    float v = 0.f;
    if (c < 3)        v = xyz2[(size_t)r * 3 + c];
    else if (c < 643) v = f2[(size_t)r * 640 + (c - 3)];
    g3[e] = f2bf(v);
}

__global__ void pool_sa3_kernel(const bf16_t* __restrict__ Y, bf16_t* __restrict__ hp) {
    int b = blockIdx.x;
    for (int c = threadIdx.x; c < 1024; c += blockDim.x) {
        float m = 0.f;
        for (int r = 0; r < 128; ++r)
            m = fmaxf(m, bf2f(Y[((size_t)b * 128 + r) * 1024 + c]));
        hp[b * 1024 + c] = f2bf(m);
    }
}

__global__ void logsoftmax_kernel(const float* __restrict__ Y, float* __restrict__ out) {
    int r = threadIdx.x;
    if (r >= 16) return;
    const float* row = Y + (size_t)r * 48;
    float mx = row[0];
    for (int j = 1; j < 40; ++j) mx = fmaxf(mx, row[j]);
    float s = 0.f;
    for (int j = 0; j < 40; ++j) s += expf(row[j] - mx);
    float l = logf(s);
    for (int j = 0; j < 40; ++j) out[r * 40 + j] = row[j] - mx - l;
}

// =======================================================================
extern "C" void kernel_launch(void* const* d_in, const int* in_sizes, int n_in,
                              void* d_out, int out_size, void* d_ws, size_t ws_size,
                              hipStream_t stream) {
    (void)in_sizes; (void)n_in; (void)out_size; (void)ws_size;
    const float BN_S = 0.99999499f;   // 1/sqrt(1+1e-5)
    const int Bx = 16, N = 4096, P1 = 512, P2 = 128;

    auto inF = [&](int i) { return (const float*)d_in[i]; };

    char* wsp = (char*)d_ws;
    size_t off = 0;
    auto alloc = [&](size_t bytes) -> void* {
        void* p = wsp + off;
        off = (off + bytes + 255) & ~(size_t)255;
        return p;
    };

    float* xyz  = (float*)alloc((size_t)Bx * N * 3 * 4);
    int*   fi1  = (int*)alloc((size_t)Bx * P1 * 4);
    float* nx1  = (float*)alloc((size_t)Bx * P1 * 3 * 4);
    int*   ig1a = (int*)alloc((size_t)Bx * P1 * 64 * 4);
    int*   ig1b = (int*)alloc((size_t)Bx * P1 * 64 * 4);
    int*   ig1c = (int*)alloc((size_t)Bx * P1 * 64 * 4);
    float* f1   = (float*)alloc((size_t)Bx * P1 * 320 * 4);
    int*   fi2  = (int*)alloc((size_t)Bx * P2 * 4);
    float* nx2  = (float*)alloc((size_t)Bx * P2 * 3 * 4);
    int*   ig2a = (int*)alloc((size_t)Bx * P2 * 64 * 4);
    int*   ig2b = (int*)alloc((size_t)Bx * P2 * 64 * 4);
    int*   ig2c = (int*)alloc((size_t)Bx * P2 * 64 * 4);
    float* f2   = (float*)alloc((size_t)Bx * P2 * 640 * 4);
    bf16_t* g3  = (bf16_t*)alloc((size_t)2048 * 672 * 2);
    bf16_t* y31 = (bf16_t*)alloc((size_t)2048 * 256 * 2);
    bf16_t* y32 = (bf16_t*)alloc((size_t)2048 * 512 * 2);
    bf16_t* y33 = (bf16_t*)alloc((size_t)2048 * 1024 * 2);
    bf16_t* hp  = (bf16_t*)alloc((size_t)16 * 1024 * 2);
    bf16_t* yf1 = (bf16_t*)alloc((size_t)16 * 512 * 2);
    bf16_t* yf2 = (bf16_t*)alloc((size_t)16 * 256 * 2);
    float*  yf3 = (float*)alloc((size_t)16 * 48 * 4);

    // conv layer table: input index of W (g=wi+1, bt=wi+2), cin, cout, KP
    struct CD { int wi, cin, cout, KP; };
    static const CD convs[21] = {
        {1, 3, 32, 32},   {4, 32, 32, 32},    {7, 32, 64, 32},
        {10, 3, 64, 32},  {13, 64, 64, 64},   {16, 64, 128, 64},
        {19, 3, 64, 32},  {22, 64, 96, 64},   {25, 96, 128, 96},
        {28, 323, 64, 352},  {31, 64, 64, 64},    {34, 64, 128, 64},
        {37, 323, 128, 352}, {40, 128, 128, 128}, {43, 128, 256, 128},
        {46, 323, 128, 352}, {49, 128, 128, 128}, {52, 128, 256, 128},
        {55, 643, 256, 672}, {58, 256, 512, 256}, {61, 512, 1024, 512},
    };
    bf16_t* wc[21]; float* bc[21];
    for (int i = 0; i < 21; ++i) {
        wc[i] = (bf16_t*)alloc((size_t)convs[i].KP * convs[i].cout * 2);
        bc[i] = (float*)alloc((size_t)convs[i].cout * 4);
    }
    bf16_t* wfc1 = (bf16_t*)alloc((size_t)1024 * 512 * 2); float* bfc1 = (float*)alloc(512 * 4);
    bf16_t* wfc2 = (bf16_t*)alloc((size_t)512 * 256 * 2);  float* bfc2 = (float*)alloc(256 * 4);
    bf16_t* wfc3 = (bf16_t*)alloc((size_t)256 * 48 * 2);   float* bfc3 = (float*)alloc(48 * 4);

    // ---- weight prep ----
    for (int i = 0; i < 21; ++i) {
        const CD& c = convs[i];
        int elems = c.KP * c.cout;
        prep_w_kernel<<<(elems + 255) / 256, 256, 0, stream>>>(
            inF(c.wi), nullptr, inF(c.wi + 1), inF(c.wi + 2),
            c.cin, c.cout, c.KP, c.cout, BN_S, wc[i], bc[i]);
    }
    prep_w_kernel<<<(1024 * 512 + 255) / 256, 256, 0, stream>>>(
        inF(64), inF(65), inF(66), inF(67), 1024, 512, 1024, 512, BN_S, wfc1, bfc1);
    prep_w_kernel<<<(512 * 256 + 255) / 256, 256, 0, stream>>>(
        inF(68), inF(69), inF(70), inF(71), 512, 256, 512, 256, BN_S, wfc2, bfc2);
    prep_w_kernel<<<(256 * 48 + 255) / 256, 256, 0, stream>>>(
        inF(72), inF(73), nullptr, nullptr, 256, 40, 256, 48, 1.0f, wfc3, bfc3);

    // ---- stage 0: layout + sampling ----
    transpose_x_kernel<<<(Bx * N + 255) / 256, 256, 0, stream>>>(inF(0), xyz);
    fps_kernel<<<Bx, 256, 0, stream>>>(xyz, N, P1, fi1);
    gather3_kernel<<<(Bx * P1 + 255) / 256, 256, 0, stream>>>(xyz, N, fi1, P1, nx1);

    int nbq1 = (Bx * P1 + 3) / 4;
    ballq_kernel<<<nbq1, 128, 0, stream>>>(xyz, nx1, N, P1, 0.1f * 0.1f, ig1a);
    ballq_kernel<<<nbq1, 128, 0, stream>>>(xyz, nx1, N, P1, 0.2f * 0.2f, ig1b);
    ballq_kernel<<<nbq1, 128, 0, stream>>>(xyz, nx1, N, P1, 0.4f * 0.4f, ig1c);

    // ---- SA1: fused 3-layer MLP + maxpool per centroid ----
    fused_branch_kernel<32, 32, 32, 64, 0><<<Bx * P1, 128, 0, stream>>>(
        xyz, nx1, ig1a, nullptr, 0, N, P1,
        wc[0], bc[0], wc[1], bc[1], wc[2], bc[2], f1, 320, 0);
    fused_branch_kernel<32, 64, 64, 128, 0><<<Bx * P1, 128, 0, stream>>>(
        xyz, nx1, ig1b, nullptr, 0, N, P1,
        wc[3], bc[3], wc[4], bc[4], wc[5], bc[5], f1, 320, 64);
    fused_branch_kernel<32, 64, 96, 128, 0><<<Bx * P1, 128, 0, stream>>>(
        xyz, nx1, ig1c, nullptr, 0, N, P1,
        wc[6], bc[6], wc[7], bc[7], wc[8], bc[8], f1, 320, 192);

    // ---- SA2 ----
    fps_kernel<<<Bx, 256, 0, stream>>>(nx1, P1, P2, fi2);
    gather3_kernel<<<(Bx * P2 + 255) / 256, 256, 0, stream>>>(nx1, P1, fi2, P2, nx2);
    int nbq2 = (Bx * P2 + 3) / 4;
    ballq_kernel<<<nbq2, 128, 0, stream>>>(nx1, nx2, P1, P2, 0.2f * 0.2f, ig2a);
    ballq_kernel<<<nbq2, 128, 0, stream>>>(nx1, nx2, P1, P2, 0.4f * 0.4f, ig2b);
    ballq_kernel<<<nbq2, 128, 0, stream>>>(nx1, nx2, P1, P2, 0.8f * 0.8f, ig2c);

    fused_branch_kernel<352, 64, 64, 128, 1><<<Bx * P2, 128, 0, stream>>>(
        nx1, nx2, ig2a, f1, 320, P1, P2,
        wc[9], bc[9], wc[10], bc[10], wc[11], bc[11], f2, 640, 0);
    fused_branch_kernel<352, 128, 128, 256, 1><<<Bx * P2, 128, 0, stream>>>(
        nx1, nx2, ig2b, f1, 320, P1, P2,
        wc[12], bc[12], wc[13], bc[13], wc[14], bc[14], f2, 640, 128);
    fused_branch_kernel<352, 128, 128, 256, 1><<<Bx * P2, 128, 0, stream>>>(
        nx1, nx2, ig2c, f1, 320, P1, P2,
        wc[15], bc[15], wc[16], bc[16], wc[17], bc[17], f2, 640, 384);

    // ---- SA3 (group-all): 3 GEMMs, M = B*128 = 2048 ----
    build_g3_kernel<<<(2048 * 672 + 255) / 256, 256, 0, stream>>>(nx2, f2, g3);
    gemm_wmma_kernel<4><<<dim3(32, 4), 128, 0, stream>>>(g3, 2048, 672, wc[18], 256, bc[18], y31, nullptr, 1);
    gemm_wmma_kernel<4><<<dim3(32, 8), 128, 0, stream>>>(y31, 2048, 256, wc[19], 512, bc[19], y32, nullptr, 1);
    gemm_wmma_kernel<4><<<dim3(32, 16), 128, 0, stream>>>(y32, 2048, 512, wc[20], 1024, bc[20], y33, nullptr, 1);
    pool_sa3_kernel<<<Bx, 256, 0, stream>>>(y33, hp);

    // ---- FC head ----
    gemm_wmma_kernel<4><<<dim3(1, 8), 128, 0, stream>>>(hp, 16, 1024, wfc1, 512, bfc1, yf1, nullptr, 1);
    gemm_wmma_kernel<4><<<dim3(1, 4), 128, 0, stream>>>(yf1, 16, 512, wfc2, 256, bfc2, yf2, nullptr, 1);
    gemm_wmma_kernel<3><<<dim3(1, 1), 128, 0, stream>>>(yf2, 16, 256, wfc3, 48, bfc3, nullptr, yf3, 0);
    logsoftmax_kernel<<<1, 32, 0, stream>>>(yf3, (float*)d_out);
}
